// DiffeqExactTraceDeepSet_49718541418572
// MI455X (gfx1250) — matrix operand order repaired and checked by reading, not compiled, not run on hardware
//
#include <hip/hip_runtime.h>
#include <hip/hip_bf16.h>

typedef __attribute__((ext_vector_type(16))) _Float16 v16h;
typedef __attribute__((ext_vector_type(8)))  _Float16 v8h;
typedef __attribute__((ext_vector_type(8)))  float    v8f;

#define DEVFN static __device__ __forceinline__

// ---------------------------------------------------------------------------
// WMMA helper: D = A(16x32 f16) * B(32x16 f16) + C(16x16 f32)
// ---------------------------------------------------------------------------
DEVFN v8f wmma16(v16h a, v16h b, v8f c) {
  return __builtin_amdgcn_wmma_f32_16x16x32_f16(
      /*neg_a=*/false, a, /*neg_b=*/false, b,
      /*c_mod=*/(short)0, c, /*reuse_a=*/false, /*reuse_b=*/false);
}

DEVFN v16h cat8(v8h lo, v8h hi) {
  return __builtin_shufflevector(lo, hi, 0, 1, 2, 3, 4, 5, 6, 7,
                                 8, 9, 10, 11, 12, 13, 14, 15);
}

// Branch-free tanh: sign(x) * (1-e)/(1+e), e = exp(-2|x|)
DEVFN float fast_tanh(float v) {
  float a = __builtin_fabsf(v);
  float e = __expf(-2.0f * a);                       // v_exp_f32
  float r = (1.0f - e) * __builtin_amdgcn_rcpf(1.0f + e);
  return __builtin_copysignf(r, v);
}

// B fragments are pre-packed per (ktile, ntile, lane): 16 contiguous halves.
DEVFN v16h load_bfrag(const _Float16* __restrict__ wp, int ntiles, int kt, int nt, int lane) {
  return *(const v16h*)(wp + (((size_t)kt * ntiles + nt) * 32 + (size_t)lane) * 16);
}

// A fragment from f16 row-major [M][K]: two 16B vector loads.
DEVFN v16h load_afrag_g(const _Float16* __restrict__ A, int K, int mrow, int kbase) {
  const v8h* p = (const v8h*)(A + (size_t)mrow * K + kbase);
  return cat8(p[0], p[2]);
}

// ---------------------------------------------------------------------------
// MADE masks (D=16, HID=256, D_H=64) -> hidden degree = (i%15)+1
// ---------------------------------------------------------------------------
DEVFN float made_mask(int mode, int k, int n) {
  if (mode == 1) {                    // input mask: in_deg[k]=k (t has deg 0)
    if (k == 0) return 1.f;
    int dn = (n % 15) + 1;
    return (dn >= k) ? 1.f : 0.f;
  }
  if (mode == 2) {                    // hidden-hidden
    int dk = (k % 15) + 1, dn = (n % 15) + 1;
    return (dn >= dk) ? 1.f : 0.f;
  }
  if (mode == 3) {                    // hidden-output (strict >)
    int dk = (k % 15) + 1, dn = (n >> 6) + 1;   // out_deg = n/64 + 1
    return (dn > dk) ? 1.f : 0.f;
  }
  return 1.f;                         // no mask (dimwise net)
}

// Pack+mask+convert weights into B-fragment order.
// out[((kt*ntiles+nt)*32 + lane)*16 + h] = mask * W[k][n],
//   k = kt*32 + 8*(lane>=16) + (h&7) + 16*(h>=8),  n = nt*16 + (lane&15)
__global__ __launch_bounds__(256) void pack_w_k(const float* __restrict__ W,
                                                _Float16* __restrict__ out,
                                                int Kreal, int Nn, int mode, int total) {
  int idx = blockIdx.x * blockDim.x + threadIdx.x;
  if (idx >= total) return;
  int h    = idx & 15;
  int lane = (idx >> 4) & 31;
  int tIdx = idx >> 9;
  int ntiles = Nn >> 4;
  int kt = tIdx / ntiles, nt = tIdx - kt * ntiles;
  int k = kt * 32 + ((lane >> 4) << 3) + (h & 7) + ((h >> 3) << 4);
  int n = nt * 16 + (lane & 15);
  float v = 0.f;
  if (k < Kreal) v = W[(size_t)k * Nn + n] * made_mask(mode, k, n);
  out[idx] = (_Float16)v;
}

// ---------------------------------------------------------------------------
// MADE layer 0: A built on the fly from [t, x] (K padded 17->32), N=256, tanh
// ---------------------------------------------------------------------------
DEVFN float l0col(float t0, const float* __restrict__ x, int row, int k, int rev) {
  if (k == 0) return t0;
  if (k <= 16) {
    int d = rev ? (16 - k) : (k - 1);
    return x[(size_t)row * 16 + d];
  }
  return 0.f;
}

__global__ __launch_bounds__(128) void made_l0_k(const float* __restrict__ t,
                                                 const float* __restrict__ x,
                                                 const _Float16* __restrict__ w0p,
                                                 const float* __restrict__ b0,
                                                 _Float16* __restrict__ out, int rev) {
  int lane = threadIdx.x & 31, wid = threadIdx.x >> 5;
  int tile = blockIdx.x * 4 + wid;          // (M/16)*16 tiles
  int mt = tile >> 4, nt = tile & 15;
  int mrow = mt * 16 + (lane & 15);
  int hi8  = (lane >> 4) << 3;
  float t0 = t[0];
  v16h a;
#pragma unroll
  for (int i = 0; i < 8; ++i) {
    a[i]     = (_Float16)l0col(t0, x, mrow, hi8 + i, rev);
    a[8 + i] = (_Float16)l0col(t0, x, mrow, hi8 + 16 + i, rev);
  }
  v16h b = load_bfrag(w0p, 16, 0, nt, lane);
  v8f c = {};
  c = wmma16(a, b, c);
  int ncol = nt * 16 + (lane & 15);
  float bb = b0[ncol];
  int rbase = mt * 16 + ((lane >> 4) << 3);
#pragma unroll
  for (int r = 0; r < 8; ++r)
    out[(size_t)(rbase + r) * 256 + ncol] = (_Float16)fast_tanh(c[r] + bb);
}

// ---------------------------------------------------------------------------
// MADE GEMM (K=256 fixed): each wave computes TWO 16x16 N-tiles sharing A.
// A (f16 row-major [M][256]) * packedW -> out f16 [M][Nn]
// ---------------------------------------------------------------------------
__global__ __launch_bounds__(128) void made_gemm_k(const _Float16* __restrict__ A,
                                                   const _Float16* __restrict__ wp,
                                                   const float* __restrict__ bias,
                                                   _Float16* __restrict__ out,
                                                   int Nn, int doTanh) {
  const int K = 256;
  int lane = threadIdx.x & 31, wid = threadIdx.x >> 5;
  int unit = blockIdx.x * 4 + wid;
  int npairs = Nn >> 5;
  int ntiles = Nn >> 4;
  int mt = unit / npairs, np = unit - mt * npairs;
  int nt0 = np * 2, nt1 = nt0 + 1;
  int mrow = mt * 16 + (lane & 15);
  int hi8  = (lane >> 4) << 3;
  v8f c0 = {}, c1 = {};
#pragma unroll
  for (int kt = 0; kt < 8; ++kt) {
    v16h a  = load_afrag_g(A, K, mrow, kt * 32 + hi8);
    v16h b0 = load_bfrag(wp, ntiles, kt, nt0, lane);
    v16h b1 = load_bfrag(wp, ntiles, kt, nt1, lane);
    c0 = wmma16(a, b0, c0);
    c1 = wmma16(a, b1, c1);
  }
  int rbase = mt * 16 + ((lane >> 4) << 3);
  int ncol0 = nt0 * 16 + (lane & 15);
  int ncol1 = nt1 * 16 + (lane & 15);
  float bb0 = bias[ncol0];
  float bb1 = bias[ncol1];
#pragma unroll
  for (int r = 0; r < 8; ++r) {
    float v0 = c0[r] + bb0;
    float v1 = c1[r] + bb1;
    if (doTanh) { v0 = fast_tanh(v0); v1 = fast_tanh(v1); }
    out[(size_t)(rbase + r) * Nn + ncol0] = (_Float16)v0;
    out[(size_t)(rbase + r) * Nn + ncol1] = (_Float16)v1;
  }
}

// ---------------------------------------------------------------------------
// Max-pool over the set dimension N of hp = p1 + rev_D(p2): mp[b][d*64+j]
// ---------------------------------------------------------------------------
__global__ __launch_bounds__(256) void maxpool_k(const _Float16* __restrict__ p1,
                                                 const _Float16* __restrict__ p2,
                                                 float* __restrict__ mp) {
  int idx = blockIdx.x * blockDim.x + threadIdx.x;   // 16*1024
  int b = idx >> 10, c = idx & 1023;
  int d = c >> 6, j = c & 63;
  int rc = (15 - d) * 64 + j;
  float m = -3.4e38f;
  for (int n = 0; n < 512; ++n) {
    size_t row = (size_t)(b * 512 + n);
    float v = (float)p1[row * 1024 + c] + (float)p2[row * 1024 + rc];
    m = fmaxf(m, v);
  }
  mp[idx] = m;
}

// h = (e1 + rev_D(e2)) + mp  ->  f16 buffer laid out as [B*N*D][64]
__global__ __launch_bounds__(256) void build_h_k(const _Float16* __restrict__ e1,
                                                 const _Float16* __restrict__ e2,
                                                 const float* __restrict__ mp,
                                                 _Float16* __restrict__ hb) {
  size_t idx = (size_t)blockIdx.x * blockDim.x + threadIdx.x;  // 8192*1024
  size_t row = idx >> 10;
  int c = (int)(idx & 1023);
  int d = c >> 6, j = c & 63;
  int rc = (15 - d) * 64 + j;
  int b = (int)(row >> 9);
  float v = (float)e1[row * 1024 + c] + (float)e2[row * 1024 + rc] + mp[b * 1024 + c];
  hb[idx] = (_Float16)v;
}

// ---------------------------------------------------------------------------
// Fused dimwise MLP + exact JVP (tangent = e_x). Each wave owns 16 rows.
//   L0: z1 = inp@W0+b0, u1=tanh, du1=(1-u1^2)*W0[1,:]   (K=96 pad, N=256)
//   L1: z2 = u1@W1+b1,  dz2 = du1@W1 (shared B frags), u2=tanh, du2=(1-u2^2)dz2
//   L2: y = u2.w2 + b2, dy = du2.w2  (per-lane dot + shfl_xor reduce)
// ---------------------------------------------------------------------------
DEVFN float dwcol(float t0, const float* __restrict__ x,
                  const _Float16* __restrict__ hb, int row, int k) {
  if (k == 0) return t0;
  if (k == 1) return x[row];
  if (k < 66) return (float)hb[(size_t)row * 64 + (k - 2)];
  return 0.f;
}

__global__ __launch_bounds__(64) void dw_fused_k(const float* __restrict__ t,
                                                 const float* __restrict__ x,
                                                 const _Float16* __restrict__ hb,
                                                 const _Float16* __restrict__ w0p,
                                                 const float* __restrict__ b0,
                                                 const float* __restrict__ dz1row,
                                                 const _Float16* __restrict__ w1p,
                                                 const float* __restrict__ b1,
                                                 const float* __restrict__ w2,
                                                 const float* __restrict__ b2,
                                                 float* __restrict__ outY,
                                                 float* __restrict__ outDY) {
  __shared__ alignas(16) _Float16 su[2][16][256];
  __shared__ alignas(16) _Float16 sdu[2][16][256];

  int lane = threadIdx.x & 31, wid = threadIdx.x >> 5;
  int rowBase = (blockIdx.x * 2 + wid) * 16;
  int mrow = rowBase + (lane & 15);
  int hi8  = (lane >> 4) << 3;
  float t0 = t[0];

  // ---- Layer 0 ----
  v16h a0[3];
#pragma unroll
  for (int kt = 0; kt < 3; ++kt) {
#pragma unroll
    for (int i = 0; i < 8; ++i) {
      a0[kt][i]     = (_Float16)dwcol(t0, x, hb, mrow, kt * 32 + hi8 + i);
      a0[kt][8 + i] = (_Float16)dwcol(t0, x, hb, mrow, kt * 32 + hi8 + 16 + i);
    }
  }
#pragma unroll 1
  for (int nt = 0; nt < 16; ++nt) {
    v8f c = {};
#pragma unroll
    for (int kt = 0; kt < 3; ++kt) {
      v16h b = load_bfrag(w0p, 16, kt, nt, lane);
      c = wmma16(a0[kt], b, c);
    }
    int ncol = nt * 16 + (lane & 15);
    float bb = b0[ncol];
    float dzv = dz1row[ncol];
    int mb = (lane >> 4) << 3;
#pragma unroll
    for (int r = 0; r < 8; ++r) {
      float u = fast_tanh(c[r] + bb);
      su[wid][mb + r][ncol]  = (_Float16)u;
      sdu[wid][mb + r][ncol] = (_Float16)((1.f - u * u) * dzv);
    }
  }
  __syncthreads();

  // ---- Layer 1 (value + tangent) fused with layer 2 dot ----
  float accY[8] = {0.f};
  float accD[8] = {0.f};
  int m = lane & 15;
#pragma unroll 1
  for (int nt = 0; nt < 16; ++nt) {
    v8f cz = {}, cd = {};
#pragma unroll 2
    for (int kt = 0; kt < 8; ++kt) {
      int k0 = kt * 32 + hi8;
      v16h au = cat8(*(const v8h*)&su[wid][m][k0],
                     *(const v8h*)&su[wid][m][k0 + 16]);
      v16h ad = cat8(*(const v8h*)&sdu[wid][m][k0],
                     *(const v8h*)&sdu[wid][m][k0 + 16]);
      v16h b = load_bfrag(w1p, 16, kt, nt, lane);
      cz = wmma16(au, b, cz);
      cd = wmma16(ad, b, cd);
    }
    int ncol = nt * 16 + (lane & 15);
    float bb = b1[ncol];
    float wv = w2[ncol];
#pragma unroll
    for (int r = 0; r < 8; ++r) {
      float u2  = fast_tanh(cz[r] + bb);
      float du2 = (1.f - u2 * u2) * cd[r];
      accY[r] += u2 * wv;
      accD[r] += du2 * wv;
    }
  }

  // reduce the 16 column-lanes of each half-wave
#pragma unroll
  for (int off = 1; off < 16; off <<= 1) {
#pragma unroll
    for (int r = 0; r < 8; ++r) {
      accY[r] += __shfl_xor(accY[r], off);
      accD[r] += __shfl_xor(accD[r], off);
    }
  }
  if ((lane & 15) == 0) {
    float bb2 = b2[0];
    int base = rowBase + ((lane >> 4) << 3);
#pragma unroll
    for (int r = 0; r < 8; ++r) {
      outY[base + r]  = accY[r] + bb2;
      outDY[base + r] = accD[r];
    }
  }
}

// ---------------------------------------------------------------------------
// Host launch
// ---------------------------------------------------------------------------
extern "C" void kernel_launch(void* const* d_in, const int* in_sizes, int n_in,
                              void* d_out, int out_size, void* d_ws, size_t ws_size,
                              hipStream_t stream) {
  (void)in_sizes; (void)out_size; (void)ws_size;
  const int M1 = 8192;      // B*N
  const int M2 = 131072;    // B*N*D

  const float* t;
  const float* x;
  const float* P[5][6];     // e1,e2,p1,p2,dimwise : W0,b0,W1,b1,W2,b2
  if (n_in >= 32) {
    t = (const float*)d_in[0];
    x = (const float*)d_in[1];
    int idx = 2;
    for (int net = 0; net < 5; ++net)
      for (int p = 0; p < 6; ++p) P[net][p] = (const float*)d_in[idx++];
  } else {
    t = (const float*)d_in[0];
    x = (const float*)d_in[1];
    const size_t madeOff[6] = {0, 4352, 4608, 70144, 70400, 332544};
    const size_t dwOff[6]   = {0, 16896, 17152, 82688, 82944, 83200};
    for (int net = 0; net < 4; ++net) {
      const float* base = (const float*)d_in[2 + net];
      for (int p = 0; p < 6; ++p) P[net][p] = base + madeOff[p];
    }
    const float* base = (const float*)d_in[6];
    for (int p = 0; p < 6; ++p) P[4][p] = base + dwOff[p];
  }

  // workspace carving (256B aligned)
  uint8_t* wsb = (uint8_t*)d_ws;
  size_t off = 0;
  auto carve = [&](size_t bytes) -> void* {
    void* p = wsb + off;
    off = (off + bytes + 255) & ~(size_t)255;
    return p;
  };
  _Float16* w0p[4]; _Float16* w1p[4]; _Float16* w2p[4];
  for (int net = 0; net < 4; ++net) {
    w0p[net] = (_Float16*)carve((size_t)32 * 256 * 2);
    w1p[net] = (_Float16*)carve((size_t)256 * 256 * 2);
    w2p[net] = (_Float16*)carve((size_t)256 * 1024 * 2);
  }
  _Float16* dw0p = (_Float16*)carve((size_t)96 * 256 * 2);
  _Float16* dw1p = (_Float16*)carve((size_t)256 * 256 * 2);
  _Float16* h1   = (_Float16*)carve((size_t)M1 * 256 * 2);
  _Float16* h2   = (_Float16*)carve((size_t)M1 * 256 * 2);
  _Float16* o[4];
  for (int net = 0; net < 4; ++net) o[net] = (_Float16*)carve((size_t)M1 * 1024 * 2);
  float*    mp = (float*)carve((size_t)16 * 1024 * 4);
  _Float16* hb = (_Float16*)carve((size_t)M1 * 1024 * 2);

  // pack masked weights into WMMA B-fragment order
  for (int net = 0; net < 4; ++net) {
    pack_w_k<<<(32 * 256) / 256, 256, 0, stream>>>(P[net][0], w0p[net], 17, 256, 1, 32 * 256);
    pack_w_k<<<(256 * 256) / 256, 256, 0, stream>>>(P[net][2], w1p[net], 256, 256, 2, 256 * 256);
    pack_w_k<<<(256 * 1024) / 256, 256, 0, stream>>>(P[net][4], w2p[net], 256, 1024, 3, 256 * 1024);
  }
  pack_w_k<<<(96 * 256) / 256, 256, 0, stream>>>(P[4][0], dw0p, 66, 256, 0, 96 * 256);
  pack_w_k<<<(256 * 256) / 256, 256, 0, stream>>>(P[4][2], dw1p, 256, 256, 0, 256 * 256);

  // four MADE nets (e2/p2 see reversed x)
  const int rev[4] = {0, 1, 0, 1};
  for (int net = 0; net < 4; ++net) {
    made_l0_k<<<(M1 / 16) * 16 / 4, 128, 0, stream>>>(t, x, w0p[net], P[net][1], h1, rev[net]);
    // 2 N-tiles per wave: units = (M/16) * (Nn/32), 4 waves per block
    made_gemm_k<<<(M1 / 16) * (256 / 32) / 4, 128, 0, stream>>>(h1, w1p[net], P[net][3], h2, 256, 1);
    made_gemm_k<<<(M1 / 16) * (1024 / 32) / 4, 128, 0, stream>>>(h2, w2p[net], P[net][5], o[net], 1024, 0);
  }

  // combine: max-pool hp over N, then h = he + mp
  maxpool_k<<<(16 * 1024) / 256, 256, 0, stream>>>(o[2], o[3], mp);
  build_h_k<<<(M1 * 1024) / 256, 256, 0, stream>>>(o[0], o[1], mp, hb);

  // fused dimwise + exact JVP -> y, dy
  dw_fused_k<<<M2 / 32, 64, 0, stream>>>(t, x, hb,
                                         dw0p, P[4][1], P[4][0] + 256,
                                         dw1p, P[4][3],
                                         P[4][4], P[4][5],
                                         (float*)d_out, (float*)d_out + M2);
}
